// SetAbstraction_34737695490780
// MI455X (gfx1250) — matrix operand order repaired and checked by reading, not compiled, not run on hardware
//
#include <hip/hip_runtime.h>

#define BATCH 4
#define NPTS  8192
#define CIN   64
#define COUT  128
#define MID   64
#define KNB   32
#define SPTS  2048
#define R2    0.01f

#define WV 4      // waves (centers) per block in sa_main
#define FP 96     // f-row pitch in ushorts (67 channels padded to 96)
#define HP 80     // h1-row pitch in ushorts (64 channels padded to 80; keeps b128 alignment)
#define OUTP (BATCH*SPTS*3)   // new_p floats preceding 'out' in d_out

typedef __attribute__((ext_vector_type(16))) __bf16 v16bf;
typedef __attribute__((ext_vector_type(8)))  float  v8f;

__device__ __forceinline__ unsigned short f2bf(float f) {
  union { float f; unsigned u; } c; c.f = f;
  unsigned u = c.u;
  u += 0x7fffu + ((u >> 16) & 1u);          // round-to-nearest-even
  return (unsigned short)(u >> 16);
}
__device__ __forceinline__ unsigned pk2(float a, float b) {
  return (unsigned)f2bf(a) | ((unsigned)f2bf(b) << 16);
}
__device__ __forceinline__ v16bf ld_frag(const unsigned short* p) {
  union { uint4 q[2]; v16bf v; } t;
  t.q[0] = *(const uint4*)p;
  t.q[1] = *(const uint4*)(p + 8);
  return t.v;
}
__device__ __forceinline__ v8f wmma_bf16(v16bf a, v16bf b, v8f c) {
  return __builtin_amdgcn_wmma_f32_16x16x32_bf16(false, a, false, b, (short)0, c,
                                                 false, false);
}

// ---------------- FPS: one block per batch; points + dist live in registers ---
__global__ __launch_bounds__(1024) void fps_kernel(const float* __restrict__ p,
                                                   int* __restrict__ idx_fps,
                                                   float* __restrict__ newp) {
  const int b   = blockIdx.x;
  const int tid = threadIdx.x;
  const float* pb = p + b * NPTS * 3;
  float px[8], py[8], pz[8], dist[8];
#pragma unroll
  for (int q = 0; q < 8; ++q) {
    int j = q * 1024 + tid;
    px[q] = pb[j*3+0]; py[q] = pb[j*3+1]; pz[q] = pb[j*3+2];
    dist[q] = 1e10f;
  }
  __shared__ float s_val[32];
  __shared__ int   s_idx[32];
  __shared__ float s_cen[3];
  __shared__ int   s_far;
  int far = 0;
  for (int i = 0; i < SPTS; ++i) {
    if (tid == 0) {
      idx_fps[b*SPTS + i] = far;
      float cx = pb[far*3+0], cy = pb[far*3+1], cz = pb[far*3+2];
      s_cen[0] = cx; s_cen[1] = cy; s_cen[2] = cz;
      newp[(b*SPTS+i)*3+0] = cx;
      newp[(b*SPTS+i)*3+1] = cy;
      newp[(b*SPTS+i)*3+2] = cz;
    }
    __syncthreads();
    float cx = s_cen[0], cy = s_cen[1], cz = s_cen[2];
    float bv = -1.f; int bi = NPTS;
#pragma unroll
    for (int q = 0; q < 8; ++q) {
      float dx = px[q]-cx, dy = py[q]-cy, dz = pz[q]-cz;
      float d = fminf(dist[q], dx*dx + dy*dy + dz*dz);
      dist[q] = d;
      int j = q*1024 + tid;
      if (d > bv || (d == bv && j < bi)) { bv = d; bi = j; }
    }
#pragma unroll
    for (int off = 16; off > 0; off >>= 1) {
      float ov = __shfl_down(bv, off);
      int   oi = __shfl_down(bi, off);
      if (ov > bv || (ov == bv && oi < bi)) { bv = ov; bi = oi; }
    }
    if ((tid & 31) == 0) { s_val[tid>>5] = bv; s_idx[tid>>5] = bi; }
    __syncthreads();
    if (tid < 32) {
      bv = s_val[tid]; bi = s_idx[tid];
#pragma unroll
      for (int off = 16; off > 0; off >>= 1) {
        float ov = __shfl_down(bv, off);
        int   oi = __shfl_down(bi, off);
        if (ov > bv || (ov == bv && oi < bi)) { bv = ov; bi = oi; }
      }
      if (tid == 0) s_far = bi;
    }
    __syncthreads();
    far = s_far;
  }
}

// ---------------- Ball query: one wave per center, ballot-ordered compaction --
__global__ __launch_bounds__(256) void ballq_kernel(const float* __restrict__ p,
                                                    const float* __restrict__ newp,
                                                    int* __restrict__ nidx) {
  const int lane = threadIdx.x & 31;
  const int cid  = (blockIdx.x * blockDim.x + threadIdx.x) >> 5;
  const int b    = cid >> 11;                 // /SPTS
  const float* pb = p + b * NPTS * 3;
  const float cx = newp[cid*3+0], cy = newp[cid*3+1], cz = newp[cid*3+2];
  int* out = nidx + cid * KNB;
  int found = 0;
  int myfirst = -1;
  for (int base = 0; base < NPTS; base += 32) {
    int j = base + lane;
    float dx = pb[j*3+0]-cx, dy = pb[j*3+1]-cy, dz = pb[j*3+2]-cz;
    bool pred = (dx*dx + dy*dy + dz*dz) < R2;
    unsigned mask = (unsigned)__ballot(pred);
    int pos = found + __popc(mask & ((1u << lane) - 1u));
    if (pred && pos < KNB) {
      out[pos] = j;
      if (pos == 0) myfirst = j;
    }
    found += __popc(mask);
    if (found >= KNB) break;
  }
#pragma unroll
  for (int off = 16; off > 0; off >>= 1) {
    int o = __shfl_xor(myfirst, off);
    myfirst = myfirst > o ? myfirst : o;
  }
  if (found < KNB && lane >= found) out[lane] = myfirst;   // pad with first idx
}

// -------- Pre-pack (gamma-folded) weights into per-lane WMMA A-fragment layout
// 16-bit A 16x32, k order chosen here (B loads match): lane L (m=L%16, h=L/16), element j -> k = 16h + j
__global__ void pack_weights(const float* __restrict__ w1, const float* __restrict__ g1,
                             const float* __restrict__ w2, const float* __restrict__ g2,
                             unsigned short* __restrict__ pw1,
                             unsigned short* __restrict__ pw2) {
  int t = blockIdx.x * blockDim.x + threadIdx.x;
  if (t < 4*3*512) {                         // stage1: 4 mtiles x 3 ktiles
    int j = t & 15, lane = (t >> 4) & 31, frag = t >> 9;
    int kt = frag % 3, mt = frag / 3;
    int h = lane >> 4, m = lane & 15;
    int k  = 16*h + j;
    int kg = kt*32 + k;
    int row = mt*16 + m;
    float v = (kg < CIN+3) ? w1[row*(CIN+3) + kg] * g1[row] : 0.f;
    pw1[t] = f2bf(v);
  }
  if (t < 8*2*512) {                         // stage2: 8 mtiles x 2 ktiles
    int j = t & 15, lane = (t >> 4) & 31, frag = t >> 9;
    int kt = frag & 1, mt = frag >> 1;
    int h = lane >> 4, m = lane & 15;
    int k = 16*h + j;
    int row = mt*16 + m;
    pw2[t] = f2bf(w2[row*MID + kt*32 + k] * g2[row]);
  }
}

// ---------------- Fused grouped MLP + max-pool + skip: one wave per center ----
__global__ __launch_bounds__(32*WV) void sa_main(
    const float* __restrict__ p, const float* __restrict__ x,
    const float* __restrict__ b1, const float* __restrict__ b2,
    const float* __restrict__ wskip, const float* __restrict__ bskip,
    const int* __restrict__ idx_fps, const int* __restrict__ nidx,
    const unsigned short* __restrict__ pw1, const unsigned short* __restrict__ pw2,
    float* __restrict__ out)
{
  __shared__ __attribute__((aligned(16))) unsigned short lds_f [WV][32*FP];
  __shared__ __attribute__((aligned(16))) unsigned short lds_h1[WV][32*HP];
  __shared__ float lds_xi[WV][CIN];
  __shared__ float lds_skip[WV][COUT];

  const int lane = threadIdx.x & 31;
  const int w    = threadIdx.x >> 5;
  const int cid  = blockIdx.x * WV + w;
  const int b    = cid >> 11;
  const int s    = cid & (SPTS-1);
  const int h    = lane >> 4, nn = lane & 15;
  const float* pb = p + b * NPTS * 3;
  const float* xb = x + b * CIN * NPTS;
  const float* newp = out;                   // fps_kernel wrote new_p there

  // ---- stage f = [dp(3) | x(64) | 0-pad] per neighbor (lane = neighbor) ----
  {
    int j = nidx[cid*KNB + lane];
    float cx = newp[cid*3+0], cy = newp[cid*3+1], cz = newp[cid*3+2];
    unsigned short* row = &lds_f[w][lane*FP];
    row[0] = f2bf(pb[j*3+0]-cx);
    row[1] = f2bf(pb[j*3+1]-cy);
    row[2] = f2bf(pb[j*3+2]-cz);
#pragma unroll 8
    for (int c = 0; c < CIN; ++c) row[3+c] = f2bf(xb[c*NPTS + j]);   // the gather
#pragma unroll
    for (int c = CIN+3; c < FP; ++c) row[c] = 0;
  }
  // ---- stage xi (center features) ----
  {
    int j0 = idx_fps[cid];
    lds_xi[w][lane]      = xb[lane*NPTS + j0];
    lds_xi[w][lane+32]   = xb[(lane+32)*NPTS + j0];
  }
  __syncthreads();

  // ---- skip GEMV: 4 outputs per lane ----
#pragma unroll
  for (int q = 0; q < 4; ++q) {
    int o = lane*4 + q;
    const float4* wr = (const float4*)(wskip + o*CIN);
    float acc = bskip[o];
#pragma unroll
    for (int c = 0; c < CIN/4; ++c) {
      float4 wv = wr[c];
      acc += wv.x*lds_xi[w][4*c]   + wv.y*lds_xi[w][4*c+1]
           + wv.z*lds_xi[w][4*c+2] + wv.w*lds_xi[w][4*c+3];
    }
    lds_skip[w][o] = acc;
  }

  // ---- stage 1: h1(64x32) = relu((g1.w1)@f + b1), bf16 WMMA, f32 acc ----
  const unsigned short* fb = &lds_f[w][0];
#pragma unroll
  for (int mt = 0; mt < 4; ++mt) {
    v8f a0 = {}; v8f a1 = {};
#pragma unroll
    for (int kt = 0; kt < 3; ++kt) {
      v16bf av  = ld_frag(pw1 + (((mt*3 + kt)*32 + lane) << 4));
      v16bf bv0 = ld_frag(fb + nn*FP      + kt*32 + 16*h);
      v16bf bv1 = ld_frag(fb + (16+nn)*FP + kt*32 + 16*h);
      a0 = wmma_bf16(av, bv0, a0);
      a1 = wmma_bf16(av, bv1, a1);
    }
    const float* bb = b1 + mt*16 + 8*h;      // rows m = 8h+r
    uint4 q0, q1;
    q0.x = pk2(fmaxf(a0[0]+bb[0],0.f), fmaxf(a0[1]+bb[1],0.f));
    q0.y = pk2(fmaxf(a0[2]+bb[2],0.f), fmaxf(a0[3]+bb[3],0.f));
    q0.z = pk2(fmaxf(a0[4]+bb[4],0.f), fmaxf(a0[5]+bb[5],0.f));
    q0.w = pk2(fmaxf(a0[6]+bb[6],0.f), fmaxf(a0[7]+bb[7],0.f));
    q1.x = pk2(fmaxf(a1[0]+bb[0],0.f), fmaxf(a1[1]+bb[1],0.f));
    q1.y = pk2(fmaxf(a1[2]+bb[2],0.f), fmaxf(a1[3]+bb[3],0.f));
    q1.z = pk2(fmaxf(a1[4]+bb[4],0.f), fmaxf(a1[5]+bb[5],0.f));
    q1.w = pk2(fmaxf(a1[6]+bb[6],0.f), fmaxf(a1[7]+bb[7],0.f));
    *(uint4*)&lds_h1[w][nn*HP      + mt*16 + 8*h] = q0;
    *(uint4*)&lds_h1[w][(16+nn)*HP + mt*16 + 8*h] = q1;
  }
  __syncthreads();

  // ---- stage 2: h2(128x32) = (g2.w2)@h1, max-pool over K, + skip, relu ----
  const unsigned short* hb = &lds_h1[w][0];
  float* ob = out + OUTP + b * COUT * SPTS + s;
#pragma unroll
  for (int mt = 0; mt < 8; ++mt) {
    v8f a0 = {}; v8f a1 = {};
#pragma unroll
    for (int kt = 0; kt < 2; ++kt) {
      v16bf av  = ld_frag(pw2 + (((mt*2 + kt)*32 + lane) << 4));
      v16bf bv0 = ld_frag(hb + nn*HP      + kt*32 + 16*h);
      v16bf bv1 = ld_frag(hb + (16+nn)*HP + kt*32 + 16*h);
      a0 = wmma_bf16(av, bv0, a0);
      a1 = wmma_bf16(av, bv1, a1);
    }
    float m[8];
#pragma unroll
    for (int r = 0; r < 8; ++r) m[r] = fmaxf(a0[r], a1[r]);
#pragma unroll
    for (int off = 1; off < 16; off <<= 1) {
#pragma unroll
      for (int r = 0; r < 8; ++r) m[r] = fmaxf(m[r], __shfl_xor(m[r], off, 16));
    }
    if (nn == 0) {                           // lanes 0 and 16 hold m=0..7 / 8..15
#pragma unroll
      for (int r = 0; r < 8; ++r) {
        int o = mt*16 + 8*h + r;
        ob[o*SPTS] = fmaxf(m[r] + b2[o] + lds_skip[w][o], 0.f);
      }
    }
  }
}

extern "C" void kernel_launch(void* const* d_in, const int* in_sizes, int n_in,
                              void* d_out, int out_size, void* d_ws, size_t ws_size,
                              hipStream_t stream) {
  const float* p     = (const float*)d_in[0];
  const float* x     = (const float*)d_in[1];
  const float* w1    = (const float*)d_in[2];
  const float* g1    = (const float*)d_in[3];
  const float* b1    = (const float*)d_in[4];
  const float* w2    = (const float*)d_in[5];
  const float* g2    = (const float*)d_in[6];
  const float* b2    = (const float*)d_in[7];
  const float* wskip = (const float*)d_in[8];
  const float* bskip = (const float*)d_in[9];
  float* out = (float*)d_out;

  // workspace layout (needs ~1.06 MB): idx_fps | nidx | packed W1 | packed W2
  char* ws = (char*)d_ws;
  int* idx_fps        = (int*)ws;                              // 32 KB
  int* nidx           = (int*)(ws + 32768);                    // 1 MB
  unsigned short* pw1 = (unsigned short*)(ws + 32768 + 1048576); // 12 KB
  unsigned short* pw2 = pw1 + 4*3*512;                         // 16 KB

  fps_kernel  <<<BATCH, 1024, 0, stream>>>(p, idx_fps, out);   // writes new_p
  pack_weights<<<32, 256, 0, stream>>>(w1, g1, w2, g2, pw1, pw2);
  ballq_kernel<<<(BATCH*SPTS*32)/256, 256, 0, stream>>>(p, out, nidx);
  sa_main     <<<(BATCH*SPTS)/WV, 32*WV, 0, stream>>>(p, x, b1, b2, wskip, bskip,
                                                      idx_fps, nidx, pw1, pw2, out);
}
